// Attention_28286654611929
// MI455X (gfx1250) — matrix operand order repaired and checked by reading, not compiled, not run on hardware
//
#include <hip/hip_runtime.h>
#include <hip/hip_bf16.h>

// ---------------------------------------------------------------------------
// Problem constants
// ---------------------------------------------------------------------------
#define BB     2      // batch
#define SS     512    // new seq len
#define DIM    2048
#define NH     32     // query heads
#define NKV    8      // kv heads
#define HD     64     // head dim
#define CL     1536   // cache len
#define LTOT   2048   // cache + new
#define MROWS  (BB*SS)   // 1024
#define NQKV   3072      // fused QKV output columns (2048 q + 512 k + 512 v)
#define GK     2048      // compile-time GEMM K (both GEMMs reduce over DIM)

typedef __attribute__((ext_vector_type(8)))  _Float16 h8;
typedef __attribute__((ext_vector_type(16))) _Float16 v16h;
typedef __attribute__((ext_vector_type(8)))  float    v8f;

// ---------------------------------------------------------------------------
// CDNA5 async global->LDS copy (ASYNCcnt-tracked), per CDNA5 ISA.
// ---------------------------------------------------------------------------
__device__ __forceinline__ void async_copy_b128(void* lds_dst, const void* gsrc) {
  unsigned ldsOff = (unsigned)(uintptr_t)lds_dst;
  asm volatile("global_load_async_to_lds_b128 %0, %1, off"
               :: "v"(ldsOff), "v"(gsrc) : "memory");
}
__device__ __forceinline__ void wait_async0() {
  asm volatile("s_wait_asynccnt 0x0" ::: "memory");
}

// ---------------------------------------------------------------------------
// WMMA operand loaders (wave32, 16x16x32 f16)
// A operand (16xK, M rows): lane L holds row M=L%16.
//   lanes 0-15 : elems 0-7 = K kk+0..7,  elems 8-15 = K kk+16..23
//   lanes16-31 : elems 0-7 = K kk+8..15, elems 8-15 = K kk+24..31
// B operand (Kx16, N cols): lane L holds col N=L%16.
//   same K order as A (lanes 0-15: kk+0..7, kk+16..23; lanes16-31: kk+8..15, kk+24..31)
// ---------------------------------------------------------------------------
__device__ __forceinline__ v16h pack16(h8 lo, h8 hi) {
  v16h r;
#pragma unroll
  for (int i = 0; i < 8; ++i) { r[i] = lo[i]; r[8 + i] = hi[i]; }
  return r;
}
__device__ __forceinline__ v16h load_A16(const _Float16* rowBase, int kk, int lane) {
  const int kb = kk + ((lane & 16) ? 8 : 0);
  h8 lo = *(const h8*)(rowBase + kb);
  h8 hi = *(const h8*)(rowBase + kb + 16);
  return pack16(lo, hi);
}
__device__ __forceinline__ v16h load_B16(const _Float16* rowBase, int kk, int lane) {
  const int kb = kk + ((lane & 16) ? 8 : 0);
  h8 lo = *(const h8*)(rowBase + kb);
  h8 hi = *(const h8*)(rowBase + kb + 16);
  return pack16(lo, hi);
}
#define WMMA_F16(a, b, c) \
  __builtin_amdgcn_wmma_f32_16x16x32_f16(false, (a), false, (b), (short)0, (c), false, false)

// ---------------------------------------------------------------------------
// GEMM: C[M,N] = A[M,K] * B[N,K]^T  (f16 in, f32 out), K = GK (compile-time).
// 32M x 64N per wave, 4 waves/block. Constant K makes every inner-loop load
// an immediate-offset b128 off 6 base pointers; the compiler unrolls and
// software-pipelines this shape cleanly (round-1 codegen).
// ---------------------------------------------------------------------------
__global__ void gemm_nt_wmma(const _Float16* __restrict__ A,
                             const _Float16* __restrict__ B,
                             float* __restrict__ Cf,
                             int M, int N) {
  const int lane = threadIdx.x & 31;
  const int wave = threadIdx.x >> 5;
  const int tile = blockIdx.x * 4 + wave;
  const int tilesN = N >> 6;                       // 64-wide N tiles
  const int tm = tile / tilesN;
  const int tn = tile - tm * tilesN;
  if (tm >= (M >> 5)) return;                      // wave-uniform

  const _Float16* arow0 = A + (size_t)(tm * 32 + (lane & 15)) * GK;
  const _Float16* arow1 = arow0 + (size_t)16 * GK;
  const _Float16* brow0 = B + (size_t)(tn * 64 + (lane & 15)) * GK;

  v8f acc[2][4];
#pragma unroll
  for (int i = 0; i < 2; ++i)
#pragma unroll
    for (int j = 0; j < 4; ++j) acc[i][j] = (v8f){};

#pragma unroll 4
  for (int kk = 0; kk < GK; kk += 32) {
    v16h a0 = load_A16(arow0, kk, lane);
    v16h a1 = load_A16(arow1, kk, lane);
#pragma unroll
    for (int j = 0; j < 4; ++j) {
      v16h b = load_B16(brow0 + (size_t)j * 16 * GK, kk, lane);
      acc[0][j] = WMMA_F16(a0, b, acc[0][j]);
      acc[1][j] = WMMA_F16(a1, b, acc[1][j]);
    }
  }

  const int halfoff = (lane & 16) ? 8 : 0;
  const int nc = lane & 15;
#pragma unroll
  for (int mi = 0; mi < 2; ++mi) {
    const int r0 = tm * 32 + mi * 16 + halfoff;
#pragma unroll
    for (int j = 0; j < 4; ++j) {
      const int col = tn * 64 + j * 16 + nc;
#pragma unroll
      for (int r = 0; r < 8; ++r) Cf[(size_t)(r0 + r) * N + col] = acc[mi][j][r];
    }
  }
}

// ---------------------------------------------------------------------------
// Elementwise helpers
// ---------------------------------------------------------------------------
__global__ void cvt_f32_to_f16(const float* __restrict__ in,
                               _Float16* __restrict__ out, int n) {
  int i = blockIdx.x * 256 + threadIdx.x;
  if (i < n) out[i] = (_Float16)in[i];
}

// qkv32 cols [0,2048) -> RoPE -> qh (b,NH,s,HD) f16
__global__ void rope_q_kernel(const float* __restrict__ qkv32,
                              const float* __restrict__ cosb,
                              const float* __restrict__ sinb,
                              _Float16* __restrict__ qh) {
  int i = blockIdx.x * 256 + threadIdx.x;          // pair index, 2^20 total
  int f  = i & 31;
  int h  = (i >> 5) & 31;
  int sr = (i >> 10) & 511;
  int b  = i >> 19;
  float c = cosb[sr * 32 + f], sn = sinb[sr * 32 + f];
  size_t src = (size_t)(b * SS + sr) * NQKV + h * HD + 2 * f;
  float xr = qkv32[src], xi = qkv32[src + 1];
  size_t dst = ((((size_t)b * NH + h) * SS + sr) * HD) + 2 * f;
  qh[dst]     = (_Float16)(xr * c - xi * sn);
  qh[dst + 1] = (_Float16)(xr * sn + xi * c);
}

// qkv32 cols [2048,2560) -> RoPE -> kh (b,NKV,LTOT,HD) f16 tail + new_k f32
__global__ void rope_k_kernel(const float* __restrict__ qkv32,
                              const float* __restrict__ cosb,
                              const float* __restrict__ sinb,
                              _Float16* __restrict__ kh,
                              float* __restrict__ newk) {
  int i = blockIdx.x * 256 + threadIdx.x;          // pair index, 2^18 total
  int f  = i & 31;
  int h  = (i >> 5) & 7;
  int sr = (i >> 8) & 511;
  int b  = i >> 17;
  float c = cosb[sr * 32 + f], sn = sinb[sr * 32 + f];
  size_t src = (size_t)(b * SS + sr) * NQKV + 2048 + h * HD + 2 * f;
  float xr = qkv32[src], xi = qkv32[src + 1];
  float outr = xr * c - xi * sn;
  float outi = xr * sn + xi * c;
  size_t kd = (((size_t)b * NKV + h) * LTOT + CL + sr) * HD + 2 * f;
  kh[kd] = (_Float16)outr; kh[kd + 1] = (_Float16)outi;
  size_t nd = (((size_t)b * NKV + h) * SS + sr) * HD + 2 * f;
  newk[nd] = outr; newk[nd + 1] = outi;
}

// qkv32 cols [2560,3072) -> vth (b,NKV,HD,LTOT) f16 tail + new_v f32
__global__ void v_pack_kernel(const float* __restrict__ qkv32,
                              _Float16* __restrict__ vth,
                              float* __restrict__ newv) {
  int i = blockIdx.x * 256 + threadIdx.x;          // element index, 2^19
  int d  = i & 63;
  int h  = (i >> 6) & 7;
  int sr = (i >> 9) & 511;
  int b  = i >> 18;
  float val = qkv32[(size_t)(b * SS + sr) * NQKV + 2560 + h * HD + d];
  vth[(((size_t)b * NKV + h) * HD + d) * LTOT + CL + sr] = (_Float16)val;
  newv[(((size_t)b * NKV + h) * SS + sr) * HD + d] = val;
}

// k_cache/v_cache (b,NKV,CL,HD) f32 -> kh rows / vth (transposed) f16
__global__ void cache_cvt_kernel(const float* __restrict__ kc,
                                 const float* __restrict__ vc,
                                 _Float16* __restrict__ kh,
                                 _Float16* __restrict__ vth) {
  int i = blockIdx.x * 256 + threadIdx.x;          // 2*8*1536*64 total
  int d  = i & 63;
  int t  = i >> 6;
  int p  = t % CL;
  int bh = t / CL;                                  // b*NKV + h
  kh[((size_t)bh * LTOT + p) * HD + d]  = (_Float16)kc[i];
  vth[((size_t)bh * HD + d) * LTOT + p] = (_Float16)vc[i];
}

// ---------------------------------------------------------------------------
// Flash attention, block-cooperative, double-buffered async K/V staging.
// Block = (b, h, half-of-s): 8 waves x 2 q-tiles share the K/V stream.
// Iteration i: wait copies issued at i-1 (into buf), barrier, issue copies
// for chunk i+1 into buf^1 (overlapped with compute), compute from buf.
// ---------------------------------------------------------------------------
__global__ void attn_wmma_kernel(const _Float16* __restrict__ qh,
                                 const _Float16* __restrict__ kh,
                                 const _Float16* __restrict__ vth,
                                 const float* __restrict__ mask,
                                 _Float16* __restrict__ aoh) {
  __shared__ _Float16 kLds[2][32][64];              // [buf][kv_local][d]  8 KB
  __shared__ _Float16 vLds[2][64][32];              // [buf][d][kv_local]  8 KB
  __shared__ _Float16 ldsP[8][2][16][32];           // per-wave P tiles   16 KB

  const int tid  = threadIdx.x;
  const int lane = tid & 31;
  const int wave = tid >> 5;
  const int sh = blockIdx.x & 1;                    // which half of s
  const int h  = (blockIdx.x >> 1) & 31;
  const int b  = blockIdx.x >> 6;
  const int kvh = h >> 2;

  const _Float16* kbase = kh  + ((size_t)b * NKV + kvh) * LTOT * HD;
  const _Float16* vbase = vth + ((size_t)b * NKV + kvh) * HD * LTOT;

  const int halfoff = (lane & 16) ? 8 : 0;
  const int nc  = lane & 15;
  const float scale = 0.125f;                       // 1/sqrt(64)

  // Q A-operands for this wave's 2 q-tiles (loaded once)
  const int qt0 = sh * 16 + wave * 2;               // global q-tile index
  v16h qa[2][2];
#pragma unroll
  for (int q = 0; q < 2; ++q) {
    const _Float16* qrowp =
        qh + ((((size_t)b * NH + h) * SS) + (qt0 + q) * 16 + nc) * HD;
    qa[q][0] = load_A16(qrowp, 0, lane);
    qa[q][1] = load_A16(qrowp, 32, lane);
  }

  float m_r[2][8], l_r[2][8];
  v8f o[2][4];
#pragma unroll
  for (int q = 0; q < 2; ++q) {
#pragma unroll
    for (int r = 0; r < 8; ++r) { m_r[q][r] = -3.0e38f; l_r[q][r] = 0.f; }
#pragma unroll
    for (int t = 0; t < 4; ++t) o[q][t] = (v8f){};
  }

  // cooperative-copy indices (256 threads): K: 32 rows x 8 16B chunks;
  //                                         V: 64 rows x 4 16B chunks
  const int kr = tid >> 3, kc = (tid & 7) * 8;
  const int vr = tid >> 2, vcc = (tid & 3) * 8;

  auto stage = [&](int buf, int j) {
    async_copy_b128(&kLds[buf][kr][kc], kbase + (size_t)(j + kr) * HD + kc);
    async_copy_b128(&vLds[buf][vr][vcc], vbase + (size_t)vr * LTOT + j + vcc);
  };

  stage(0, 0);                                      // pipeline prologue

  for (int j0 = 0; j0 < LTOT; j0 += 32) {
    const int buf = (j0 >> 5) & 1;
    wait_async0();                                  // my copies into buf done
    __syncthreads();                                // everyone's done; prev reads done
    if (j0 + 32 < LTOT) stage(buf ^ 1, j0 + 32);    // overlap next DMA w/ compute

    // ---- S = Q K^T for both q-tiles (K B-operand shared) -------------------
    v8f sacc[2][2];
#pragma unroll
    for (int t2 = 0; t2 < 2; ++t2) {
      const _Float16* krow = &kLds[buf][t2 * 16 + nc][0];
      v16h kb0 = load_B16(krow, 0, lane);
      v16h kb1 = load_B16(krow, 32, lane);
#pragma unroll
      for (int q = 0; q < 2; ++q) {
        v8f s = (v8f){};
        s = WMMA_F16(qa[q][0], kb0, s);
        s = WMMA_F16(qa[q][1], kb1, s);
        sacc[q][t2] = s;
      }
    }

    // ---- online softmax + write P tiles ------------------------------------
#pragma unroll
    for (int q = 0; q < 2; ++q) {
      const int qr0 = (qt0 + q) * 16 + halfoff;
#pragma unroll
      for (int r = 0; r < 8; ++r) {
        const float* mrow = mask + (size_t)(qr0 + r) * LTOT + j0;
        float s0 = sacc[q][0][r] * scale + mrow[nc];
        float s1 = sacc[q][1][r] * scale + mrow[16 + nc];
        float mx = fmaxf(s0, s1);
#pragma unroll
        for (int off = 8; off; off >>= 1) mx = fmaxf(mx, __shfl_xor(mx, off, 32));
        float mnew = fmaxf(m_r[q][r], mx);
        float sf = __expf(m_r[q][r] - mnew);
        m_r[q][r] = mnew;
        float p0 = __expf(s0 - mnew);
        float p1 = __expf(s1 - mnew);
        float rs = p0 + p1;
#pragma unroll
        for (int off = 8; off; off >>= 1) rs += __shfl_xor(rs, off, 32);
        l_r[q][r] = l_r[q][r] * sf + rs;
#pragma unroll
        for (int t = 0; t < 4; ++t) o[q][t][r] *= sf;
        ldsP[wave][q][halfoff + r][nc]      = (_Float16)p0;
        ldsP[wave][q][halfoff + r][16 + nc] = (_Float16)p1;
      }
    }

    // ---- V B-operands (shared across q-tiles) + PV -------------------------
    v16h vb[4];
#pragma unroll
    for (int t = 0; t < 4; ++t)
      vb[t] = load_B16(&vLds[buf][t * 16 + nc][0], 0, lane);

    const int cb = (lane & 16) ? 8 : 0;
#pragma unroll
    for (int q = 0; q < 2; ++q) {
      h8 plo = *(const h8*)&ldsP[wave][q][nc][cb];
      h8 phi = *(const h8*)&ldsP[wave][q][nc][cb + 16];
      v16h pa = pack16(plo, phi);
#pragma unroll
      for (int t = 0; t < 4; ++t) o[q][t] = WMMA_F16(pa, vb[t], o[q][t]);
    }
  }

  // ---- normalize + store f16 to aoh (b, s, NH*HD) --------------------------
#pragma unroll
  for (int q = 0; q < 2; ++q) {
#pragma unroll
    for (int r = 0; r < 8; ++r) {
      float inv = 1.f / l_r[q][r];
      size_t row = (size_t)b * SS + (qt0 + q) * 16 + halfoff + r;
#pragma unroll
      for (int t = 0; t < 4; ++t)
        aoh[row * (NH * HD) + h * HD + t * 16 + nc] = (_Float16)(o[q][t][r] * inv);
    }
  }
}

// ---------------------------------------------------------------------------
// Host orchestration
// ---------------------------------------------------------------------------
extern "C" void kernel_launch(void* const* d_in, const int* in_sizes, int n_in,
                              void* d_out, int out_size, void* d_ws, size_t ws_size,
                              hipStream_t stream) {
  const float* x     = (const float*)d_in[0];
  const float* fcos  = (const float*)d_in[1];
  const float* fsin  = (const float*)d_in[2];
  const float* kc    = (const float*)d_in[3];
  const float* vc    = (const float*)d_in[4];
  const float* amask = (const float*)d_in[5];
  const float* wq    = (const float*)d_in[6];
  const float* wk    = (const float*)d_in[7];
  const float* wv    = (const float*)d_in[8];
  const float* wo    = (const float*)d_in[9];

  char* ws = (char*)d_ws;
  size_t off = 0;
  _Float16* xh   = (_Float16*)(ws + off); off += (size_t)MROWS * DIM * 2;
  _Float16* wqh  = (_Float16*)(ws + off); off += (size_t)DIM * DIM * 2;       // |
  _Float16* wkh  = (_Float16*)(ws + off); off += (size_t)(NKV*HD) * DIM * 2;  // | contiguous:
  _Float16* wvh  = (_Float16*)(ws + off); off += (size_t)(NKV*HD) * DIM * 2;  // | fused QKV B
  _Float16* woh  = (_Float16*)(ws + off); off += (size_t)DIM * DIM * 2;
  float*    qkv32= (float*)(ws + off);    off += (size_t)MROWS * NQKV * 4;    // 12 MB
  _Float16* qqh  = (_Float16*)(ws + off); off += (size_t)BB*NH*SS*HD * 2;
  _Float16* kh   = (_Float16*)(ws + off); off += (size_t)BB*NKV*LTOT*HD * 2;
  _Float16* vth  = (_Float16*)(ws + off); off += (size_t)BB*NKV*HD*LTOT * 2;
  _Float16* aoh  = (_Float16*)(ws + off); off += (size_t)MROWS * DIM * 2;

  float* out_o = (float*)d_out;                       // (b,s,dim)
  float* out_k = out_o + (size_t)BB * SS * DIM;       // (b,NKV,s,HD)
  float* out_v = out_k + (size_t)BB * NKV * SS * HD;  // (b,NKV,s,HD)

  // 1) f32 -> f16 casts (wq/wk/wv land contiguously => single fused B matrix)
  cvt_f32_to_f16<<<(MROWS*DIM)/256, 256, 0, stream>>>(x,  xh,  MROWS*DIM);
  cvt_f32_to_f16<<<(DIM*DIM)/256,   256, 0, stream>>>(wq, wqh, DIM*DIM);
  cvt_f32_to_f16<<<(NKV*HD*DIM)/256,256, 0, stream>>>(wk, wkh, NKV*HD*DIM);
  cvt_f32_to_f16<<<(NKV*HD*DIM)/256,256, 0, stream>>>(wv, wvh, NKV*HD*DIM);
  cvt_f32_to_f16<<<(DIM*DIM)/256,   256, 0, stream>>>(wo, woh, DIM*DIM);

  // 2) fused QKV projection: [x] @ [wq;wk;wv]^T -> qkv32 (1024 x 3072)
  gemm_nt_wmma<<<(MROWS/32)*(NQKV/64)/4, 128, 0, stream>>>(
      xh, wqh, qkv32, MROWS, NQKV);

  // 3) RoPE + KV repack (+ new_k/new_v outputs) + cache cast
  rope_q_kernel<<<(BB*SS*NH*32)/256, 256, 0, stream>>>(qkv32, fcos, fsin, qqh);
  rope_k_kernel<<<(BB*SS*NKV*32)/256, 256, 0, stream>>>(qkv32, fcos, fsin, kh, out_k);
  v_pack_kernel<<<(BB*SS*NKV*HD)/256, 256, 0, stream>>>(qkv32, vth, out_v);
  cache_cvt_kernel<<<(BB*NKV*CL*HD)/256, 256, 0, stream>>>(kc, vc, kh, vth);

  // 4) flash attention: 128 blocks x 8 waves (block = b,h,s-half)
  attn_wmma_kernel<<<BB*NH*2, 256, 0, stream>>>(qqh, kh, vth, amask, aoh);

  // 5) output projection -> d_out (f32)
  gemm_nt_wmma<<<(MROWS/32)*(DIM/64)/4, 128, 0, stream>>>(
      aoh, woh, out_o, MROWS, DIM);
}
